// Model_16612933501123
// MI455X (gfx1250) — matrix operand order repaired and checked by reading, not compiled, not run on hardware
//
#include <hip/hip_runtime.h>
#include <math.h>

// ---------------- problem constants ----------------
constexpr int B_  = 64;
constexpr int T_  = 13;
constexpr int N_  = 325;
constexpr int H_  = 64;
constexpr int C_  = 4;
constexpr int L_  = 2;
constexpr int S_  = 4;
constexpr int P_  = 12;
constexpr int NC_  = S_ * N_;      // 1300 combined nodes
constexpr int NCP_ = 1312;         // K padded to multiple of 32 (41 * 32)
constexpr int BH_  = B_ * H_;      // 4096 feature columns
constexpr int HH_  = H_ * H_;      // 4096
constexpr int CH_  = C_ * H_;      // 256

typedef __attribute__((ext_vector_type(16))) __bf16 v16bf;
typedef __attribute__((ext_vector_type(8)))  float  v8f;

struct Pack { uint4 a, b; };   // 32 bytes == v16bf
union FragU { v16bf v; unsigned short u[16]; };

__device__ __forceinline__ unsigned short f2bf(float f) {
  unsigned int u = __float_as_uint(f);
  u += 0x7fffu + ((u >> 16) & 1u);   // round-to-nearest-even
  return (unsigned short)(u >> 16);
}

// ---------------- adjacency fp32 -> bf16, K-padded lda (once per launch) ----------------
__global__ __launch_bounds__(256) void k_cvt_adj(
    const float* __restrict__ a0, const float* __restrict__ a1,
    const float* __restrict__ a2, const float* __restrict__ a3,
    unsigned short* __restrict__ dst)
{
  size_t per = (size_t)NC_ * NCP_;
  size_t i = blockIdx.x * (size_t)256 + threadIdx.x;
  if (i >= 4 * per) return;
  int which = (int)(i / per);
  size_t rem = i % per;
  int r = (int)(rem / NCP_);
  int c = (int)(rem % NCP_);
  const float* src = (which == 0) ? a0 : (which == 1) ? a1 : (which == 2) ? a2 : a3;
  dst[i] = (c < NC_) ? f2bf(src[(size_t)r * NC_ + c]) : (unsigned short)0;
}

// ---------------- zero K-pad rows of Hmat + 4 H1 slabs (once per launch) ----------------
__global__ __launch_bounds__(256) void k_zero_pad(unsigned short* __restrict__ Hmat,
                                                  unsigned short* __restrict__ H1)
{
  const int padElems = (NCP_ - NC_) * BH_;              // per slab
  int i = blockIdx.x * 256 + threadIdx.x;
  if (i >= padElems * (1 + C_)) return;
  int slab = i / padElems;
  int off  = i % padElems;
  size_t idx = (size_t)NC_ * BH_ + off;
  if (slab == 0) Hmat[idx] = 0;
  else           H1[(size_t)(slab - 1) * NCP_ * BH_ + idx] = 0;
}

// ---------------- x = inputs @ w_in + b_in  [B,T,N,H] ----------------
__global__ __launch_bounds__(256) void k_input_proj(
    const float* __restrict__ in, const float* __restrict__ w_in,
    const float* __restrict__ b_in, float* __restrict__ x)
{
  size_t i = blockIdx.x * (size_t)256 + threadIdx.x;
  if (i >= (size_t)B_ * T_ * N_ * H_) return;
  int h = (int)(i & 63);
  size_t btn = i >> 6;
  x[i] = in[btn * 2 + 0] * w_in[h] + in[btn * 2 + 1] * w_in[64 + h] + b_in[h];
}

// ---------------- build Hmat(bf16 [NCP,BH]) + gf + residual ----------------
__global__ __launch_bounds__(256) void k_build_feats(
    const float* __restrict__ x, const float* __restrict__ last,
    unsigned short* __restrict__ Hmat, float* __restrict__ gf,
    float* __restrict__ resid,
    const float* __restrict__ gconv_w, const float* __restrict__ gconv_b,
    const float* __restrict__ res_w, const float* __restrict__ res_b,
    int idx, int left)
{
  int i = blockIdx.x * 256 + threadIdx.x;       // over B*N*H
  if (i >= B_ * N_ * H_) return;
  int h  = i & 63;
  int bn = i >> 6;
  int n  = bn % N_;
  int b  = bn / N_;
  float gacc = 0.f, racc = 0.f;
  for (int s = 0; s < S_; ++s) {
    float v;
    if (idx == 0)      v = x[(size_t)((b * T_ + s) * N_ + n) * H_ + h];
    else if (s == 0)   v = last[(size_t)bn * H_ + h];
    else               v = x[(size_t)((b * T_ + (left + s - 1)) * N_ + n) * H_ + h];
    Hmat[(size_t)(s * N_ + n) * BH_ + b * H_ + h] = f2bf(v);
    gacc += gconv_w[s] * v;
    racc += res_w[s] * v;
  }
  gf[(size_t)bn * H_ + h]    = gacc + gconv_b[0];
  resid[(size_t)bn * H_ + h] = racc + res_b[0];
}

// ---------------- WMMA bf16 GEMM:  C[f32, M x 4096] = A[bf16,M x KP] @ B[bf16,KP x 4096]
// block = 256 thr = 8 waves; block tile 128x256; wave tile 64x64 (4x4 wmma accs)
// KP is a multiple of 32 (zero-padded) -> no K guards, fully vectorized staging.
#define BM 128
#define BN 256
#define BK 32
#define LDA_SH 40    // ushort stride: 80 B rows (16B aligned)
#define LDB_SH 264   // ushort stride: 528 B rows (16B aligned)

__global__ __launch_bounds__(256) void k_msg_gemm(
    const unsigned short* __restrict__ A, int lda, int rowOff, int Mrows,
    const unsigned short* __restrict__ Bm, int KP,
    float* __restrict__ Cc)
{
  __shared__ __align__(16) unsigned short sA[BM * LDA_SH];   // [m][k]
  __shared__ __align__(16) unsigned short sB[BK * LDB_SH];   // [k][n] row-major

  const int tid    = threadIdx.x;
  const int lane   = tid & 31;
  const int laneLo = lane & 15;
  const int laneHi = lane >> 4;
  const int w  = tid >> 5;
  const int wy = w & 1;        // M half (0..1)
  const int wx = w >> 1;       // N quarter (0..3)
  const int mtile = blockIdx.x * BM;
  const int ntile = blockIdx.y * BN;

  v8f acc[4][4];
  for (int i = 0; i < 4; ++i)
    for (int j = 0; j < 4; ++j)
      for (int r = 0; r < 8; ++r) acc[i][j][r] = 0.f;

  for (int k0 = 0; k0 < KP; k0 += BK) {
    // stage A tile [128 x 32] : 2 aligned uint4 chunks per thread
#pragma unroll
    for (int it = 0; it < 2; ++it) {
      int cidx = tid + it * 256;          // 0..511
      int m  = cidx >> 2;
      int kc = (cidx & 3) * 8;
      int gm = mtile + m;
      uint4 v = make_uint4(0u, 0u, 0u, 0u);
      if (gm < Mrows)
        v = *reinterpret_cast<const uint4*>(&A[(size_t)(rowOff + gm) * lda + k0 + kc]);
      *reinterpret_cast<uint4*>(&sA[m * LDA_SH + kc]) = v;
    }
    // stage B tile [32 x 256] row-major : 4 aligned uint4 chunks per thread
#pragma unroll
    for (int it = 0; it < 4; ++it) {
      int cidx = tid + it * 256;          // 0..1023
      int k  = cidx >> 5;
      int nc = (cidx & 31) * 8;
      uint4 v = *reinterpret_cast<const uint4*>(&Bm[(size_t)(k0 + k) * BH_ + ntile + nc]);
      *reinterpret_cast<uint4*>(&sB[k * LDB_SH + nc]) = v;
    }
    if (k0 + BK < KP)
      __builtin_prefetch(&Bm[(size_t)(k0 + BK) * BH_ + ntile + ((tid & 31) * 8)], 0, 0);
    __syncthreads();

    // A frags: lanes 0-15 -> K {0..7,16..23}; lanes 16-31 -> K {8..15,24..31}
    v16bf afrag[4];
#pragma unroll
    for (int mi = 0; mi < 4; ++mi) {
      int row = wy * 64 + mi * 16 + laneLo;
      const unsigned short* p = &sA[row * LDA_SH + laneHi * 8];
      Pack fr;
      fr.a = *reinterpret_cast<const uint4*>(p);
      fr.b = *reinterpret_cast<const uint4*>(p + 16);
      afrag[mi] = __builtin_bit_cast(v16bf, fr);
    }
    // B frags: per-lane col = laneLo, 16 K values (transpose gather from row-major tile)
    v16bf bfrag[4];
#pragma unroll
    for (int ni = 0; ni < 4; ++ni) {
      int col = wx * 64 + ni * 16 + laneLo;
      FragU fu;
#pragma unroll
      for (int i = 0; i < 16; ++i)
        fu.u[i] = sB[(laneHi * 8 + i + (i & 8)) * LDB_SH + col];
      bfrag[ni] = fu.v;
    }
#pragma unroll
    for (int mi = 0; mi < 4; ++mi)
#pragma unroll
      for (int ni = 0; ni < 4; ++ni)
        acc[mi][ni] = __builtin_amdgcn_wmma_f32_16x16x32_bf16(
            false, afrag[mi], false, bfrag[ni], (short)0, acc[mi][ni], false, false);
    __syncthreads();
  }

  // store: v8f reg r -> M = laneHi*8 + r, N = laneLo
#pragma unroll
  for (int mi = 0; mi < 4; ++mi) {
    int rowBase = wy * 64 + mi * 16 + laneHi * 8;
#pragma unroll
    for (int ni = 0; ni < 4; ++ni) {
      int col = ntile + wx * 64 + ni * 16 + laneLo;
#pragma unroll
      for (int r = 0; r < 8; ++r) {
        int gm = mtile + rowBase + r;
        if (gm < Mrows) Cc[(size_t)gm * BH_ + col] = acc[mi][ni][r];
      }
    }
  }
}

// ---------------- per-node H transform: relu(MF@W0 + MB@W1 + bias) ----------------
__global__ __launch_bounds__(256) void k_hmm(
    const float* __restrict__ MF, const float* __restrict__ MB,
    const float* __restrict__ W0, const float* __restrict__ W1,
    const float* __restrict__ bias, int Mm,
    unsigned short* __restrict__ out_bf, float* __restrict__ out_f32, int outStride)
{
  __shared__ float xs[4][2][64];
  int grp = threadIdx.x >> 6;
  int h2  = threadIdx.x & 63;
  int pair = blockIdx.x * 4 + grp;
  bool valid = pair < Mm * B_;
  int m = valid ? pair / B_ : 0;
  int b = valid ? pair % B_ : 0;
  size_t base = (size_t)m * BH_ + b * H_;
  if (valid) { xs[grp][0][h2] = MF[base + h2]; xs[grp][1][h2] = MB[base + h2]; }
  __syncthreads();
  if (!valid) return;
  float acc = bias[h2];
  for (int h = 0; h < 64; ++h)
    acc += xs[grp][0][h] * W0[h * 64 + h2] + xs[grp][1][h] * W1[h * 64 + h2];
  float y = fmaxf(acc, 0.f);
  if (out_bf)  out_bf[base + h2] = f2bf(y);
  if (out_f32) out_f32[(size_t)(b * N_ + m) * outStride + h2] = y;
}

// ---------------- gate + blend + reduce + residual ----------------
__global__ __launch_bounds__(256) void k_gate_reduce(
    const float* __restrict__ gf, const float* __restrict__ ca,
    const float* __restrict__ cp, const float* __restrict__ resid,
    const float* __restrict__ g1w, const float* __restrict__ g1b,
    const float* __restrict__ g2w, const float* __restrict__ g2b,
    const float* __restrict__ rw,  const float* __restrict__ rb,
    float* __restrict__ cr)
{
  __shared__ float gfr[64], t1[64], gca[256];
  int row = blockIdx.x;                 // (b*N + n)
  int tid = threadIdx.x;
  if (tid < 64) gfr[tid] = gf[(size_t)row * 64 + tid];
  __syncthreads();
  if (tid < 64) {
    float s = g1b[tid];
    for (int h = 0; h < 64; ++h) s += gfr[h] * g1w[h * 64 + tid];
    t1[tid] = fmaxf(s, 0.f);
  }
  __syncthreads();
  {
    float z = g2b[tid];
    for (int h = 0; h < 64; ++h) z += t1[h] * g2w[h * 256 + tid];
    float g = 1.f / (1.f + __expf(-z));
    size_t o = (size_t)row * 256 + tid;
    gca[tid] = g * ca[o] + (1.f - g) * cp[o];
  }
  __syncthreads();
  if (tid < 64) {
    float s = rb[tid];
    for (int j = 0; j < 256; ++j) s += gca[j] * rw[j * 64 + tid];
    cr[(size_t)row * 64 + tid] = s + resid[(size_t)row * 64 + tid];
  }
}

// ---------------- global LayerNorm: 2-stage deterministic reduction ----------------
__global__ __launch_bounds__(256) void k_reduce_partial(
    const float* __restrict__ cr, float* __restrict__ part, int n)
{
  __shared__ float ss[256], sq[256];
  float s = 0.f, q = 0.f;
  for (int i = blockIdx.x * 256 + threadIdx.x; i < n; i += gridDim.x * 256) {
    float v = cr[i]; s += v; q += v * v;
  }
  ss[threadIdx.x] = s; sq[threadIdx.x] = q;
  __syncthreads();
  for (int o = 128; o > 0; o >>= 1) {
    if (threadIdx.x < o) { ss[threadIdx.x] += ss[threadIdx.x + o]; sq[threadIdx.x] += sq[threadIdx.x + o]; }
    __syncthreads();
  }
  if (threadIdx.x == 0) { part[2 * blockIdx.x] = ss[0]; part[2 * blockIdx.x + 1] = sq[0]; }
}

__global__ __launch_bounds__(256) void k_reduce_final(
    const float* __restrict__ part, int nb, float inv_n, float* __restrict__ stats)
{
  __shared__ float ss[256], sq[256];
  float s = 0.f, q = 0.f;
  for (int i = threadIdx.x; i < nb; i += 256) { s += part[2 * i]; q += part[2 * i + 1]; }
  ss[threadIdx.x] = s; sq[threadIdx.x] = q;
  __syncthreads();
  for (int o = 128; o > 0; o >>= 1) {
    if (threadIdx.x < o) { ss[threadIdx.x] += ss[threadIdx.x + o]; sq[threadIdx.x] += sq[threadIdx.x + o]; }
    __syncthreads();
  }
  if (threadIdx.x == 0) {
    float mu  = ss[0] * inv_n;
    float var = sq[0] * inv_n - mu * mu;
    stats[0] = mu;
    stats[1] = rsqrtf(var + 1e-5f);
  }
}

__global__ __launch_bounds__(256) void k_norm(
    const float* __restrict__ cr, const float* __restrict__ stats,
    float* __restrict__ last, int n)
{
  int i = blockIdx.x * 256 + threadIdx.x;
  if (i >= n) return;
  last[i] = (cr[i] - stats[0]) * stats[1];
}

// ---------------- output head ----------------
__global__ __launch_bounds__(256) void k_output(
    const float* __restrict__ last, const float* __restrict__ ocw,
    const float* __restrict__ ocb, const float* __restrict__ olw,
    const float* __restrict__ olb, float* __restrict__ out)
{
  int i = blockIdx.x * 256 + threadIdx.x;     // over B*P*N
  if (i >= B_ * P_ * N_) return;
  int n = i % N_;
  int t = i / N_;
  int p = t % P_;
  int b = t / P_;
  float w = ocw[p], bb = ocb[p];
  const float* lp = last + (size_t)(b * N_ + n) * H_;
  float acc = 0.f;
  for (int h = 0; h < 64; ++h) acc += fmaxf(w * lp[h] + bb, 0.f) * olw[h];
  out[i] = acc + olb[0];
}

// ================= host orchestration =================
extern "C" void kernel_launch(void* const* d_in, const int* in_sizes, int n_in,
                              void* d_out, int out_size, void* d_ws, size_t ws_size,
                              hipStream_t stream) {
  const float* inp     = (const float*)d_in[0];
  const float* adjF    = (const float*)d_in[1];
  const float* adjB    = (const float*)d_in[2];
  const float* peaF    = (const float*)d_in[3];
  const float* peaB    = (const float*)d_in[4];
  const float* w_in    = (const float*)d_in[5];
  const float* b_in    = (const float*)d_in[6];
  const float* res_w   = (const float*)d_in[7];
  const float* res_b   = (const float*)d_in[8];
  const float* gconv_w = (const float*)d_in[9];
  const float* gconv_b = (const float*)d_in[10];
  const float* g1w     = (const float*)d_in[11];
  const float* g1b     = (const float*)d_in[12];
  const float* g2w     = (const float*)d_in[13];
  const float* g2b     = (const float*)d_in[14];
  const float* rw      = (const float*)d_in[15];
  const float* rb      = (const float*)d_in[16];
  const float* WA      = (const float*)d_in[17];  // [C,L,2,H,H]
  const float* bA      = (const float*)d_in[18];  // [C,L,H]
  const float* WP      = (const float*)d_in[19];
  const float* bP      = (const float*)d_in[20];
  const float* ocw     = (const float*)d_in[21];
  const float* ocb     = (const float*)d_in[22];
  const float* olw     = (const float*)d_in[23];
  const float* olb     = (const float*)d_in[24];
  float* out = (float*)d_out;

  // ---- workspace carve-up ----
  char* wp = (char*)d_ws;
  auto alloc = [&](size_t bytes) -> char* {
    char* r = wp;
    wp += (bytes + 255) & ~(size_t)255;
    return r;
  };
  unsigned short* Abf   = (unsigned short*)alloc((size_t)4 * NC_ * NCP_ * 2);
  float*          xbuf  = (float*)alloc((size_t)B_ * T_ * N_ * H_ * 4);
  float*          last  = (float*)alloc((size_t)B_ * N_ * H_ * 4);
  unsigned short* Hmat  = (unsigned short*)alloc((size_t)NCP_ * BH_ * 2);
  float*          MF    = (float*)alloc((size_t)NC_ * BH_ * 4);
  float*          MB    = (float*)alloc((size_t)NC_ * BH_ * 4);
  unsigned short* H1    = (unsigned short*)alloc((size_t)C_ * NCP_ * BH_ * 2);
  float*          ca    = (float*)alloc((size_t)B_ * N_ * CH_ * 4);
  float*          cp    = (float*)alloc((size_t)B_ * N_ * CH_ * 4);
  float*          gf    = (float*)alloc((size_t)B_ * N_ * H_ * 4);
  float*          resid = (float*)alloc((size_t)B_ * N_ * H_ * 4);
  float*          cr    = (float*)alloc((size_t)B_ * N_ * H_ * 4);
  float*          part  = (float*)alloc(512 * 2 * 4);
  float*          stats = (float*)alloc(2 * 4);

  // ---- one-time per launch ----
  {
    size_t tot = (size_t)4 * NC_ * NCP_;
    k_cvt_adj<<<(unsigned)((tot + 255) / 256), 256, 0, stream>>>(adjF, adjB, peaF, peaB, Abf);
  }
  {
    int tot = (NCP_ - NC_) * BH_ * (1 + C_);
    k_zero_pad<<<(tot + 255) / 256, 256, 0, stream>>>(Hmat, H1);
  }
  {
    size_t tot = (size_t)B_ * T_ * N_ * H_;
    k_input_proj<<<(unsigned)((tot + 255) / 256), 256, 0, stream>>>(inp, w_in, b_in, xbuf);
  }

  const int cps[4] = {4, 7, 10, 13};
  const int Mtot = B_ * N_ * H_;
  int left = 0;
  for (int idx = 0; idx < 4; ++idx) {
    k_build_feats<<<(Mtot + 255) / 256, 256, 0, stream>>>(
        xbuf, last, Hmat, gf, resid, gconv_w, gconv_b, res_w, res_b, idx, left);

    for (int type = 0; type < 2; ++type) {
      const unsigned short* Af = Abf + (size_t)(type == 0 ? 0 : 2) * NC_ * NCP_;
      const unsigned short* Ab = Af + (size_t)NC_ * NCP_;
      const float* W    = (type == 0) ? WA : WP;
      const float* Bv   = (type == 0) ? bA : bP;
      float* outCat     = (type == 0) ? ca : cp;

      // ----- layer 0: shared messages across channels -----
      dim3 g0((NC_ + BM - 1) / BM, BH_ / BN);
      k_msg_gemm<<<g0, 256, 0, stream>>>(Af, NCP_, 0, NC_, Hmat, NCP_, MF);
      k_msg_gemm<<<g0, 256, 0, stream>>>(Ab, NCP_, 0, NC_, Hmat, NCP_, MB);
      for (int c = 0; c < C_; ++c) {
        k_hmm<<<(NC_ * B_ + 3) / 4, 256, 0, stream>>>(
            MF, MB,
            W + (size_t)((c * L_ + 0) * 2 + 0) * HH_,
            W + (size_t)((c * L_ + 0) * 2 + 1) * HH_,
            Bv + (size_t)(c * L_ + 0) * H_,
            NC_, H1 + (size_t)c * NCP_ * BH_, nullptr, 0);
      }
      // ----- layer 1: only last-snapshot rows needed -----
      dim3 g1((N_ + BM - 1) / BM, BH_ / BN);
      for (int c = 0; c < C_; ++c) {
        const unsigned short* Hc = H1 + (size_t)c * NCP_ * BH_;
        k_msg_gemm<<<g1, 256, 0, stream>>>(Af, NCP_, 3 * N_, N_, Hc, NCP_, MF);
        k_msg_gemm<<<g1, 256, 0, stream>>>(Ab, NCP_, 3 * N_, N_, Hc, NCP_, MB);
        k_hmm<<<(N_ * B_ + 3) / 4, 256, 0, stream>>>(
            MF, MB,
            W + (size_t)((c * L_ + 1) * 2 + 0) * HH_,
            W + (size_t)((c * L_ + 1) * 2 + 1) * HH_,
            Bv + (size_t)(c * L_ + 1) * H_,
            N_, nullptr, outCat + c * H_, CH_);
      }
    }

    k_gate_reduce<<<B_ * N_, 256, 0, stream>>>(
        gf, ca, cp, resid, g1w, g1b, g2w, g2b, rw, rb, cr);

    k_reduce_partial<<<512, 256, 0, stream>>>(cr, part, Mtot);
    k_reduce_final<<<1, 256, 0, stream>>>(part, 512, 1.0f / (float)Mtot, stats);
    k_norm<<<(Mtot + 255) / 256, 256, 0, stream>>>(cr, stats, last, Mtot);

    left = cps[idx];
  }

  k_output<<<(B_ * P_ * N_ + 255) / 256, 256, 0, stream>>>(last, ocw, ocb, olw, olb, out);
}
